// WorkingLearnableTensorSketch_21818433864280
// MI455X (gfx1250) — compile-verified, matrix operand order checked by reading, not run on hardware
//
#include <hip/hip_runtime.h>
#include <hip/hip_bf16.h>

typedef __attribute__((ext_vector_type(16))) _Float16 v16h;
typedef __attribute__((ext_vector_type(8)))  float    v8f;

#define NBLK 64
#define NTHR 256
#define REC_STRIDE 16   // floats per per-block record in workspace

// Chunk transform of the collapsed affine recurrence:
//   a' = a + p ; b' = q*a + b + r ; c' = u*a + s*b + c + w
struct Tup { float p, q, r, u, s, w; };

__device__ __forceinline__ Tup compose(const Tup& A, const Tup& B) {
  // Apply A first, then B (order matters — non-commutative).
  Tup o;
  o.p = A.p + B.p;
  o.q = A.q + B.q;
  o.s = A.s + B.s;
  o.r = A.r + B.r + B.q * A.p;
  o.u = A.u + B.u + B.s * A.q;
  o.w = A.w + B.w + B.u * A.p + B.s * A.r;
  return o;
}

// Compose accumulated transform t with a single-character leaf (p2,q2,0,0,s2,0).
// Uses pre-update p,q,r (write order matters).
__device__ __forceinline__ void step_char(Tup& t, unsigned& pack, int c,
                                          const float* l_p, const float* l_q,
                                          const float* l_s) {
  const float q2 = l_q[c], s2 = l_s[c];
  t.w += s2 * t.r;
  t.u += s2 * t.q;
  t.r += q2 * t.p;
  t.p += l_p[c];
  t.q += q2;
  t.s += s2;
  pack += 1u << (8 * c);
}

__global__ void sketch_reduce_kernel(const int* __restrict__ seq,
                                     const float* __restrict__ sign_table,
                                     float* __restrict__ ws, int n) {
  __shared__ float sp[NTHR], sq[NTHR], sr[NTHR], su[NTHR], ss[NTHR], sw[NTHR];
  __shared__ unsigned scnt[4];
  const int tid = threadIdx.x;
  if (tid < 4) scnt[tid] = 0u;

  // Per-character leaf values: p=s0/D, q=s1, s=s2 (hash_table is a no-op on
  // uniform vectors and drops out of the math entirely).
  float l_p[4], l_q[4], l_s[4];
  #pragma unroll
  for (int c = 0; c < 4; ++c) {
    l_p[c] = sign_table[c] * (1.0f / 128.0f);
    l_q[c] = sign_table[4 + c];
    l_s[c] = sign_table[8 + c];
  }

  const int tot = NBLK * NTHR;
  const int cpt = (n + tot - 1) / tot;           // chars per thread (8 for SEQ=131072)
  const int gid = blockIdx.x * NTHR + tid;
  const int base = gid * cpt;                    // contiguous, order-preserving chunks

  Tup t = {0.f, 0.f, 0.f, 0.f, 0.f, 0.f};
  unsigned pack = 0u;                            // 4 x 8-bit histogram (cpt <= 255)

  if (cpt == 8 && base + 8 <= n) {
    // Fast path (exact fit, SEQ=131072): two global_load_b128 per thread,
    // fully unrolled 8-char compose chain. base is a multiple of 8 ->
    // 32-byte-aligned 128-bit loads.
    const int4* v = (const int4*)(seq + base);
    const int4 v0 = v[0];
    const int4 v1 = v[1];
    step_char(t, pack, v0.x & 3, l_p, l_q, l_s);
    step_char(t, pack, v0.y & 3, l_p, l_q, l_s);
    step_char(t, pack, v0.z & 3, l_p, l_q, l_s);
    step_char(t, pack, v0.w & 3, l_p, l_q, l_s);
    step_char(t, pack, v1.x & 3, l_p, l_q, l_s);
    step_char(t, pack, v1.y & 3, l_p, l_q, l_s);
    step_char(t, pack, v1.z & 3, l_p, l_q, l_s);
    step_char(t, pack, v1.w & 3, l_p, l_q, l_s);
  } else {
    for (int j = 0; j < cpt; ++j) {
      const int idx = base + j;
      if (idx < n) step_char(t, pack, seq[idx] & 3, l_p, l_q, l_s);
    }
  }

  sp[tid] = t.p; sq[tid] = t.q; sr[tid] = t.r;
  su[tid] = t.u; ss[tid] = t.s; sw[tid] = t.w;
  __syncthreads();

  atomicAdd(&scnt[0],  pack        & 0xFFu);
  atomicAdd(&scnt[1], (pack >> 8)  & 0xFFu);
  atomicAdd(&scnt[2], (pack >> 16) & 0xFFu);
  atomicAdd(&scnt[3], (pack >> 24) & 0xFFu);

  // Ordered (adjacent-pair) tree reduction across the block.
  for (int stride = 1; stride < NTHR; stride <<= 1) {
    __syncthreads();
    if ((tid & (2 * stride - 1)) == 0) {
      const int o = tid + stride;
      Tup A = { sp[tid], sq[tid], sr[tid], su[tid], ss[tid], sw[tid] };
      Tup B = { sp[o],   sq[o],   sr[o],   su[o],   ss[o],   sw[o]   };
      Tup R = compose(A, B);
      sp[tid] = R.p; sq[tid] = R.q; sr[tid] = R.r;
      su[tid] = R.u; ss[tid] = R.s; sw[tid] = R.w;
    }
  }
  __syncthreads();

  if (tid == 0) {
    float* rec = ws + (size_t)blockIdx.x * REC_STRIDE;
    rec[0] = sp[0]; rec[1] = sq[0]; rec[2] = sr[0];
    rec[3] = su[0]; rec[4] = ss[0]; rec[5] = sw[0];
    rec[6] = (float)scnt[0]; rec[7] = (float)scnt[1];
    rec[8] = (float)scnt[2]; rec[9] = (float)scnt[3];
  }
}

__global__ void sketch_finalize_kernel(const float* __restrict__ ws,
                                       const float* __restrict__ gscale,
                                       const float* __restrict__ cscales,
                                       const float* __restrict__ dweights,
                                       const float* __restrict__ cpert,
                                       float* __restrict__ out, int n) {
  __shared__ float sBase;
  __shared__ float sProb[4];
  const int tid = threadIdx.x;

  if (tid == 0) {
    // Ordered serial combine of the 64 block records (trivial work).
    Tup acc = {0.f, 0.f, 0.f, 0.f, 0.f, 0.f};
    float cnt0 = 0.f, cnt1 = 0.f, cnt2 = 0.f, cnt3 = 0.f;
    for (int b = 0; b < NBLK; ++b) {
      const float* rec = ws + (size_t)b * REC_STRIDE;
      Tup B = { rec[0], rec[1], rec[2], rec[3], rec[4], rec[5] };
      acc = compose(acc, B);
      cnt0 += rec[6]; cnt1 += rec[7]; cnt2 += rec[8]; cnt3 += rec[9];
    }
    const float inv_n = 1.0f / (float)n;
    const float pr0 = cnt0 * inv_n, pr1 = cnt1 * inv_n,
                pr2 = cnt2 * inv_n, pr3 = cnt3 * inv_n;
    sProb[0] = pr0; sProb[1] = pr1; sProb[2] = pr2; sProb[3] = pr3;
    const float scaling = 0.9f + 0.2f * (pr0 * cscales[0] + pr1 * cscales[1] +
                                         pr2 * cscales[2] + pr3 * cscales[3]);
    const float baseline = acc.w * inv_n;        // T[3] is uniform: value = w_total / n
    sBase = baseline * gscale[0] * scaling;
  }
  __syncthreads();

  const float base = sBase;
  const float p0 = sProb[0], p1 = sProb[1], p2 = sProb[2], p3 = sProb[3];

  // probs(1x4) @ char_perturbations(4x128) via v_wmma_f32_16x16x32_f16:
  // 8 waves, one 16-column tile each; A row 0 = probs (K=0..3), rest zero.
  const int lane = tid & 31;
  const int wv   = tid >> 5;               // wave id 0..7
  const int col  = wv * 16 + (lane & 15);  // output column this lane covers
  const bool lo  = (lane < 16);
  const bool l0  = (lane == 0);            // holds A row M=0, K=0..3

  v16h a = {};
  a[0] = l0 ? (_Float16)p0 : (_Float16)0.f;
  a[1] = l0 ? (_Float16)p1 : (_Float16)0.f;
  a[2] = l0 ? (_Float16)p2 : (_Float16)0.f;
  a[3] = l0 ? (_Float16)p3 : (_Float16)0.f;

  const float P0 = cpert[0 * 128 + col];
  const float P1 = cpert[1 * 128 + col];
  const float P2 = cpert[2 * 128 + col];
  const float P3 = cpert[3 * 128 + col];
  v16h b = {};
  b[0] = lo ? (_Float16)P0 : (_Float16)0.f;
  b[1] = lo ? (_Float16)P1 : (_Float16)0.f;
  b[2] = lo ? (_Float16)P2 : (_Float16)0.f;
  b[3] = lo ? (_Float16)P3 : (_Float16)0.f;

  v8f c = {};
  // (neg_a, A, neg_b, B, c_mod, C, reuse_a, reuse_b) -> v_wmma_f32_16x16x32_f16
  c = __builtin_amdgcn_wmma_f32_16x16x32_f16(false, a, false, b, (short)0, c,
                                             false, false);

  if (lo) {
    // D VGPR0, lanes 0..15 hold row M=0 (the probs row-product).
    out[col] = base * dweights[col] + 0.1f * c[0];
  }
}

extern "C" void kernel_launch(void* const* d_in, const int* in_sizes, int n_in,
                              void* d_out, int out_size, void* d_ws, size_t ws_size,
                              hipStream_t stream) {
  const int*   seq = (const int*)d_in[0];
  // d_in[1] (hash_table) is provably unused: rolls act on uniform vectors.
  const float* sgn = (const float*)d_in[2];
  const float* gs  = (const float*)d_in[3];
  const float* cs  = (const float*)d_in[4];
  const float* dw  = (const float*)d_in[5];
  const float* cp  = (const float*)d_in[6];
  float* out = (float*)d_out;
  float* ws  = (float*)d_ws;     // needs NBLK*REC_STRIDE*4 = 4 KB
  const int n = in_sizes[0];

  sketch_reduce_kernel<<<NBLK, NTHR, 0, stream>>>(seq, sgn, ws, n);
  sketch_finalize_kernel<<<1, NTHR, 0, stream>>>(ws, gs, cs, dw, cp, out, n);
}
